// MACE_36825049596259
// MI455X (gfx1250) — matrix operand order repaired and checked by reading, not compiled, not run on hardware
//
#include <hip/hip_runtime.h>
#include <hip/hip_bf16.h>
#include <math.h>

typedef __attribute__((ext_vector_type(16))) _Float16 v16h;
typedef __attribute__((ext_vector_type(8)))  float    v8f;

#define N_NODES 8000
#define N_EDGES 128000
#define ZSP 10
#define KF 128
#define WAVES 8

#define WMMA_F16(A, B, C) \
  __builtin_amdgcn_wmma_f32_16x16x32_f16(false, (A), false, (B), (short)0, (C), false, false)

// fast silu: x * v_rcp_f32(1 + exp(-x))  (no IEEE div chain)
__device__ __forceinline__ float silu_f(float x) {
  return x * __builtin_amdgcn_rcpf(1.f + __expf(-x));
}

// Load one pre-swizzled fragment (32 lanes x 16 halves, contiguous, 32B/lane)
__device__ __forceinline__ v16h load_frag(const _Float16* base, int frag, int lane) {
  return *(const v16h*)(base + ((frag << 5) + lane) * 16);
}

// Scatter C (16x16 f32) into A-fragment layout with silu applied.
// C layout: lane holds col n = nbase + lane%16, rows m = i + 8*(lane>=16).
// A-frag inverse map for element (m,k): frag = k>>5, lane' = m + 16*((k>>4)&1),
// j' = k&15.
__device__ __forceinline__ void store_silu_frag(_Float16* actf, int nbase, int lane, v8f c) {
  const int n   = nbase + (lane & 15);       // == k for the next layer
  const int mh  = (lane >> 4) << 3;
  const int kb  = n >> 5;
  const int jp  = n & 15;
  const int lhi = ((n >> 4) & 1) << 4;
#pragma unroll
  for (int i = 0; i < 8; ++i)
    actf[(kb << 9) + (mh + i + lhi) * 16 + jp] = (_Float16)silu_f(c[i]);
}

// ---------------------------------------------------------------------------
// Prep (1 block): fold species-dependent algebra into PW/E0, and emit the MLP
// weights in WMMA B-fragment order (fragment-major, 32 lanes x 16 halves).
//   frag id = nb*nkb + kb ; n = nb*16 + lane%16 ; k = kb*32 + (lane/16)*16 + j
// ---------------------------------------------------------------------------
__global__ __launch_bounds__(256) void mace_prep(
    const float* __restrict__ atomic_energies, const float* __restrict__ w_embed,
    const float* __restrict__ w_up,    const float* __restrict__ W1,
    const float* __restrict__ W2,      const float* __restrict__ W3,
    const float* __restrict__ W4,      const float* __restrict__ w_lin,
    const float* __restrict__ w_skip,  const float* __restrict__ w_sym,
    const float* __restrict__ w_lin2,  const float* __restrict__ w_readout,
    float* __restrict__ PW, float* __restrict__ E0, _Float16* __restrict__ Wf)
{
  __shared__ float s_wl2r[KF];
  __shared__ float s_HU[ZSP * KF];
  __shared__ float s_Q[ZSP * KF];
  const int tid = threadIdx.x;

  for (int k = tid; k < KF; k += 256) {
    float acc = 0.f;
    for (int j = 0; j < KF; ++j) acc += w_lin2[k * KF + j] * w_readout[j];
    s_wl2r[k] = acc;
  }
  __syncthreads();
  for (int idx = tid; idx < ZSP * KF; idx += 256) {
    const int z = idx / KF, k = idx % KF;
    float hu = 0.f, q = 0.f;
    for (int t = 0; t < KF; ++t) hu += w_embed[z * KF + t] * w_up[t * KF + k];
    for (int j = 0; j < KF; ++j) q += w_lin[k * KF + j] * w_sym[z * KF + j] * s_wl2r[j];
    s_HU[idx] = hu;
    s_Q[idx]  = q;
  }
  __syncthreads();
  for (int idx = tid; idx < ZSP * ZSP * KF; idx += 256) {
    const int k = idx % KF, zr = (idx / KF) % ZSP, zs = idx / (KF * ZSP);
    PW[idx] = s_HU[zs * KF + k] * s_Q[zr * KF + k];
  }
  for (int z = tid; z < ZSP; z += 256) {
    float acc = 0.f;
    for (int k = 0; k < KF; ++k) {
      float w = 0.f;
      for (int j = 0; j < KF; ++j) w += w_skip[(z * KF + k) * KF + j] * w_readout[j];
      acc += w_embed[z * KF + k] * w;
    }
    E0[z] = atomic_energies[z] + acc * 0.31622776601f; // 1/sqrt(Z)
  }

  // fragment-ordered weights, packed contiguously:
  _Float16* W1f = Wf;          // 4 frags  (nb=0..3, kb=0, rows 8..31 zero) : 2048
  _Float16* W2f = Wf + 2048;   // 8 frags  (nb*2+kb)                        : 4096
  _Float16* W3f = Wf + 6144;   // 8 frags                                   : 4096
  _Float16* W4f = Wf + 10240;  // 16 frags (nb*2+kb), col 4n of W4          : 8192
  for (int i = tid; i < 4 * 512; i += 256) {
    const int j = i & 15, lane = (i >> 4) & 31, nb = i >> 9;
    const int n = nb * 16 + (lane & 15);
    const int k = ((lane >> 4) << 4) + j;
    W1f[i] = (k < 8) ? (_Float16)W1[k * 64 + n] : (_Float16)0.f;
  }
  for (int i = tid; i < 8 * 512; i += 256) {
    const int j = i & 15, lane = (i >> 4) & 31, f = i >> 9;
    const int n = (f >> 1) * 16 + (lane & 15);
    const int k = (f & 1) * 32 + ((lane >> 4) << 4) + j;
    W2f[i] = (_Float16)W2[k * 64 + n];
    W3f[i] = (_Float16)W3[k * 64 + n];
  }
  for (int i = tid; i < 16 * 512; i += 256) {
    const int j = i & 15, lane = (i >> 4) & 31, f = i >> 9;
    const int n = (f >> 1) * 16 + (lane & 15);
    const int k = (f & 1) * 32 + ((lane >> 4) << 4) + j;
    W4f[i] = (_Float16)W4[k * 512 + 4 * n];   // column 4n = tpw[:,:,l=0]
  }
}

__global__ void mace_init(const int* __restrict__ species, const float* __restrict__ E0,
                          float* __restrict__ out) {
  const int i = blockIdx.x * blockDim.x + threadIdx.x;
  if (i < N_NODES) out[i] = E0[species[i]];
}

// ---------------------------------------------------------------------------
// Edge kernel: one wave per 16-edge tile.
// ---------------------------------------------------------------------------
__global__ __launch_bounds__(256) void mace_edges(
    const float* __restrict__ pos,   const float* __restrict__ shifts,
    const int*   __restrict__ eidx,  const int*   __restrict__ species,
    const float* __restrict__ PW,    const _Float16* __restrict__ gWf,
    float* __restrict__ out)
{
  __shared__ __align__(32) _Float16 sW[18432];            // all weight fragments
  __shared__ __align__(32) _Float16 sActF[WAVES][1024];   // 2 A-fragments per wave
  __shared__ int sRecv[WAVES][16];
  __shared__ int sPWB[WAVES][16];

  const int tid = threadIdx.x;
  {  // streaming 16B copy of all weight fragments into LDS
    const uint4* src = (const uint4*)gWf;
    uint4* dst = (uint4*)sW;
    for (int i = tid; i < 18432 * 2 / 16; i += 256) dst[i] = src[i];
  }
  __syncthreads();
  const _Float16* sW1f = sW;
  const _Float16* sW2f = sW + 2048;
  const _Float16* sW3f = sW + 6144;
  const _Float16* sW4f = sW + 10240;

  const int wave = tid >> 5, lane = tid & 31;
  const int eb = (blockIdx.x * WAVES + wave) << 4;
  _Float16* actf = sActF[wave];

  // ---- geometry: build the layer-1 A-fragment directly in registers.
  // Lane L<16 holds edge m=L with k=j (j<8) and k=8..15 (zero) ; lanes>=16 all zero.
  v16h a0 = {};
  if (lane < 16) {
    const int e  = eb + lane;
    const int si = eidx[e];
    const int ri = eidx[N_EDGES + e];
    const float dx = pos[ri * 3 + 0] - pos[si * 3 + 0] + shifts[e * 3 + 0];
    const float dy = pos[ri * 3 + 1] - pos[si * 3 + 1] + shifts[e * 3 + 1];
    const float dz = pos[ri * 3 + 2] - pos[si * 3 + 2] + shifts[e * 3 + 2];
    const float r = sqrtf(dx * dx + dy * dy + dz * dz + 1e-9f);
    const float x = r * 0.2f;                       // r / R_MAX
    float env = 0.f;
    if (x < 1.f) {
      const float x5 = x * x * x * x * x;
      env = 1.f - 21.f * x5 + 35.f * x5 * x - 15.f * x5 * x * x;  // P=5 envelope
    }
    const float pref = 0.6324555320f * env * __builtin_amdgcn_rcpf(r);
    const float arg  = 0.6283185307f * r;           // pi * r / R_MAX
#pragma unroll
    for (int n = 1; n <= 8; ++n)
      a0[n - 1] = (_Float16)(pref * __sinf((float)n * arg));
    sRecv[wave][lane] = ri;
    sPWB[wave][lane]  = (species[si] * ZSP + species[ri]) * KF;
  }

  // ---- layer 1: (16x32 padded) @ (32x64)
#pragma unroll
  for (int nb = 0; nb < 4; ++nb) {
    v8f c = {};
    c = WMMA_F16(a0, load_frag(sW1f, nb, lane), c);
    store_silu_frag(actf, nb * 16, lane, c);
  }
  asm volatile("s_wait_dscnt 0" ::: "memory");

  // ---- layer 2
  a0 = *(const v16h*)(actf + lane * 16);
  v16h a1 = *(const v16h*)(actf + 512 + lane * 16);
#pragma unroll
  for (int nb = 0; nb < 4; ++nb) {
    v8f c = {};
    c = WMMA_F16(a0, load_frag(sW2f, nb * 2 + 0, lane), c);
    c = WMMA_F16(a1, load_frag(sW2f, nb * 2 + 1, lane), c);
    store_silu_frag(actf, nb * 16, lane, c);
  }
  asm volatile("s_wait_dscnt 0" ::: "memory");

  // ---- layer 3
  a0 = *(const v16h*)(actf + lane * 16);
  a1 = *(const v16h*)(actf + 512 + lane * 16);
#pragma unroll
  for (int nb = 0; nb < 4; ++nb) {
    v8f c = {};
    c = WMMA_F16(a0, load_frag(sW3f, nb * 2 + 0, lane), c);
    c = WMMA_F16(a1, load_frag(sW3f, nb * 2 + 1, lane), c);
    store_silu_frag(actf, nb * 16, lane, c);
  }
  asm volatile("s_wait_dscnt 0" ::: "memory");

  // ---- layer 4: (16x64) @ (64x128) fused with PW dot
  a0 = *(const v16h*)(actf + lane * 16);
  a1 = *(const v16h*)(actf + 512 + lane * 16);
  const int mh = (lane >> 4) << 3;
  int pwb[8];
#pragma unroll
  for (int i = 0; i < 8; ++i) pwb[i] = sPWB[wave][mh + i];
  float g[8] = {0.f, 0.f, 0.f, 0.f, 0.f, 0.f, 0.f, 0.f};
#pragma unroll
  for (int nb = 0; nb < 8; ++nb) {
    v8f c = {};
    c = WMMA_F16(a0, load_frag(sW4f, nb * 2 + 0, lane), c);
    c = WMMA_F16(a1, load_frag(sW4f, nb * 2 + 1, lane), c);
    const int k = nb * 16 + (lane & 15);   // C col n -> feature index k
#pragma unroll
    for (int i = 0; i < 8; ++i) g[i] += PW[pwb[i] + k] * c[i];   // row m = mh+i
  }
  // reduce across the 16 lanes holding the same edge row (masks 1..8 stay in-half)
#pragma unroll
  for (int i = 0; i < 8; ++i) {
    float v = g[i];
    v += __shfl_xor(v, 1, 32);
    v += __shfl_xor(v, 2, 32);
    v += __shfl_xor(v, 4, 32);
    v += __shfl_xor(v, 8, 32);
    g[i] = v;
  }
  if ((lane & 15) == 0) {
#pragma unroll
    for (int i = 0; i < 8; ++i)
      atomicAdd(&out[sRecv[wave][mh + i]], g[i] * 0.0625f);   // / AVG_NEIGH
  }
}

extern "C" void kernel_launch(void* const* d_in, const int* in_sizes, int n_in,
                              void* d_out, int out_size, void* d_ws, size_t ws_size,
                              hipStream_t stream) {
  const float* positions       = (const float*)d_in[0];
  const float* shifts          = (const float*)d_in[1];
  const int*   edge_index      = (const int*)  d_in[2];
  const int*   species         = (const int*)  d_in[3];
  const float* atomic_energies = (const float*)d_in[4];
  const float* w_embed         = (const float*)d_in[5];
  const float* w_up            = (const float*)d_in[6];
  const float* W1              = (const float*)d_in[7];
  const float* W2              = (const float*)d_in[8];
  const float* W3              = (const float*)d_in[9];
  const float* W4              = (const float*)d_in[10];
  const float* w_lin           = (const float*)d_in[11];
  const float* w_skip          = (const float*)d_in[12];
  const float* w_sym           = (const float*)d_in[13];
  const float* w_lin2          = (const float*)d_in[14];
  const float* w_readout       = (const float*)d_in[15];
  float* out = (float*)d_out;

  // workspace layout (~88 KB total)
  char* ws = (char*)d_ws;
  float*    PW = (float*)   (ws + 0);       // 12800 f32 = 51200 B
  float*    E0 = (float*)   (ws + 51200);   // 10 f32
  _Float16* Wf = (_Float16*)(ws + 51456);   // 18432 f16 fragments = 36864 B -> end 88320

  mace_prep<<<1, 256, 0, stream>>>(atomic_energies, w_embed, w_up, W1, W2, W3, W4,
                                   w_lin, w_skip, w_sym, w_lin2, w_readout, PW, E0, Wf);
  mace_init<<<(N_NODES + 255) / 256, 256, 0, stream>>>(species, E0, out);
  mace_edges<<<(N_EDGES / 16) / WAVES, 256, 0, stream>>>(positions, shifts, edge_index,
                                                         species, PW, Wf, out);
}
